// AdaptiveAlphaQuantizedLinear_64330020159884
// MI455X (gfx1250) — matrix operand order repaired and checked by reading, not compiled, not run on hardware
//
#include <hip/hip_runtime.h>

// ---------------------------------------------------------------------------
// AdaptiveAlphaQuantizedLinear on MI455X (gfx1250)
//   out[m, k] = sum_n x[m,n] * (Wq[k,n]-zero[k,g])*scale[k,g]*mu2[k]*mu1[n] + bias[k]
//   M=256, K=8192, N(red)=8192, group=128 along n.
// HBM-bound on the 256 MB int32 Wq stream (~11.7 us floor @ 23.3 TB/s);
// bf16 WMMA (f32 accum) keeps compute well under that floor.
// Workgroup: 128m x 64ko, 8 waves of 32m x 32ko; 64-n reduction step,
// double-buffered LDS staging of dequantized W, 2x-unrolled pipeline.
// ---------------------------------------------------------------------------

typedef __attribute__((ext_vector_type(16))) __bf16 v16bf;
typedef __attribute__((ext_vector_type(8)))  float  v8f;

#define M_ROWS   256
#define K_OUT    8192
#define N_RED    8192
#define NG       64      // n-groups (group size 128 along n)
#define NSTEP    64      // reduction elements staged per pipeline step
#define LDS_PITCH 144    // bytes per ko-row (64 bf16 + pad): 16B aligned,
                         // col*36 mod 64 banks all distinct -> conflict-free b128

// ---------------- x: fp32 -> bf16 (one-time, 2M elements) ----------------
__global__ __launch_bounds__(256) void xcvt_kernel(const float* __restrict__ x,
                                                   unsigned short* __restrict__ xb) {
    int i = (blockIdx.x * 256 + threadIdx.x) * 8;
    float4 a = *(const float4*)(x + i);
    float4 b = *(const float4*)(x + i + 4);
    union { __bf16 h[8]; uint4 u; } p;
    p.h[0] = (__bf16)a.x; p.h[1] = (__bf16)a.y;
    p.h[2] = (__bf16)a.z; p.h[3] = (__bf16)a.w;
    p.h[4] = (__bf16)b.x; p.h[5] = (__bf16)b.y;
    p.h[6] = (__bf16)b.z; p.h[7] = (__bf16)b.w;
    *(uint4*)(xb + i) = p.u;
}

// ---------------- main GEMM: dequant -> LDS -> bf16 WMMA ----------------
__global__ __launch_bounds__(256) void qlin_kernel(
    const unsigned short* __restrict__ xb,    // [256, 8192] bf16
    const int*            __restrict__ Wq,    // [8192, 8192] int32 codes 0..15
    const float*          __restrict__ scales,// [K, NG]
    const float*          __restrict__ zeros, // [K, NG]
    const float*          __restrict__ mu1,   // [N]
    const float*          __restrict__ mu2,   // [K]
    const float*          __restrict__ bias,  // [K]
    float*                __restrict__ out)   // [256, 8192]
{
    __shared__ __align__(16) unsigned char ldsB[2][64 * LDS_PITCH];

    const int tid  = threadIdx.x;
    const int lane = tid & 31;
    const int wave = tid >> 5;

    const int ko_blk = blockIdx.x * 64;    // 64 output columns per workgroup
    const int m_blk  = blockIdx.y * 128;   // 128 output rows per workgroup

    // ---- staging role: thread t dequantizes 16 codes of row (ko_blk + t/4) ----
    const int   s_ko   = tid >> 2;                 // 0..63
    const int   s_n    = (tid & 3) * 16;           // 0,16,32,48
    const long  wrow   = (long)(ko_blk + s_ko) * N_RED;
    const float mu2k   = mu2[ko_blk + s_ko];
    const float* sc_row = scales + (ko_blk + s_ko) * NG;
    const float* zp_row = zeros  + (ko_blk + s_ko) * NG;

    // ---- compute role: wave tile 32m x 32ko (2x2 subtiles of 16x16) ----
    const int m_wave  = m_blk  + (wave >> 1) * 32;
    const int ko_wave = ko_blk + (wave & 1) * 32;
    const int a_koff  = (lane >> 4) * 16;          // A layout: lanes 16-31 hold K+16
    const int a_m0    = m_wave + (lane & 15);
    const int b_boff  = (lane >> 4) * 32;          // B layout: lanes 16-31 hold K+16
    const int b_col0  = (wave & 1) * 32 + (lane & 15);

    v8f acc[2][2] = {};

    // stage: dequantize 64ko x 64n slab of W into LDS buffer `buf`
    auto stage = [&](int buf, int nb) {
        const int   g  = nb >> 7;                  // 64-chunk never splits a group
        const float z  = zp_row[g];
        const float cs = sc_row[g] * mu2k;
        const int*   qp = Wq  + wrow + nb + s_n;
        const float* up = mu1 + nb + s_n;
        unsigned char* dst = &ldsB[buf][s_ko * LDS_PITCH + s_n * 2];
        #pragma unroll
        for (int h = 0; h < 2; ++h) {              // 2 x 8 elements
            const int4   q0 = *(const int4*)(qp + h * 8);
            const int4   q1 = *(const int4*)(qp + h * 8 + 4);
            const float4 u0 = *(const float4*)(up + h * 8);
            const float4 u1 = *(const float4*)(up + h * 8 + 4);
            union { __bf16 hh[8]; uint4 u; } p;
            p.hh[0] = (__bf16)(((float)q0.x - z) * (cs * u0.x));
            p.hh[1] = (__bf16)(((float)q0.y - z) * (cs * u0.y));
            p.hh[2] = (__bf16)(((float)q0.z - z) * (cs * u0.z));
            p.hh[3] = (__bf16)(((float)q0.w - z) * (cs * u0.w));
            p.hh[4] = (__bf16)(((float)q1.x - z) * (cs * u1.x));
            p.hh[5] = (__bf16)(((float)q1.y - z) * (cs * u1.y));
            p.hh[6] = (__bf16)(((float)q1.z - z) * (cs * u1.z));
            p.hh[7] = (__bf16)(((float)q1.w - z) * (cs * u1.w));
            *(uint4*)(dst + h * 16) = p.u;
        }
        // keep the HBM W stream two steps ahead (global_prefetch_b8)
        __builtin_prefetch(qp + 2 * NSTEP, 0, 0);
    };

    // compute: 8 WMMA over a staged 64-n slab
    auto compute = [&](int buf, int nb) {
        union F { uint4 u[2]; v16bf v; };
        #pragma unroll
        for (int kc = 0; kc < 2; ++kc) {           // two 32-n chunks
            F bfr[2];
            #pragma unroll
            for (int c = 0; c < 2; ++c) {          // B frag: 32 contiguous LDS bytes/lane
                const unsigned char* p =
                    &ldsB[buf][(b_col0 + c * 16) * LDS_PITCH + kc * 64 + b_boff];
                bfr[c].u[0] = *(const uint4*)p;
                bfr[c].u[1] = *(const uint4*)(p + 16);
            }
            #pragma unroll
            for (int ms = 0; ms < 2; ++ms) {       // A frag: two 16B global loads
                const unsigned short* ap =
                    xb + (long)(a_m0 + ms * 16) * N_RED + nb + kc * 32 + a_koff;
                F afr;
                afr.u[0] = *(const uint4*)ap;
                afr.u[1] = *(const uint4*)(ap + 8);
                #pragma unroll
                for (int c = 0; c < 2; ++c)
                    acc[ms][c] = __builtin_amdgcn_wmma_f32_16x16x32_bf16(
                        false, afr.v, false, bfr[c].v, (short)0, acc[ms][c],
                        false, false);
            }
        }
    };

    // ---- double-buffered pipeline, 2x unrolled so buffer idx is constant ----
    stage(0, 0);
    __syncthreads();
    int nb = 0;
    for (; nb < N_RED - 2 * NSTEP; nb += 2 * NSTEP) {
        stage(1, nb + NSTEP);
        compute(0, nb);
        __syncthreads();
        stage(0, nb + 2 * NSTEP);
        compute(1, nb + NSTEP);
        __syncthreads();
    }
    // tail: nb == N_RED - 2*NSTEP (N_RED/NSTEP = 128, even)
    stage(1, nb + NSTEP);
    compute(0, nb);
    __syncthreads();
    compute(1, nb + NSTEP);

    // ---- epilogue: C/D layout -> out[m,k] + bias ----
    #pragma unroll
    for (int ms = 0; ms < 2; ++ms) {
        #pragma unroll
        for (int c = 0; c < 2; ++c) {
            const int k    = ko_wave + c * 16 + (lane & 15);
            const float bv = bias[k];
            const int mrow = m_wave + ms * 16 + (lane >> 4) * 8;
            #pragma unroll
            for (int r = 0; r < 8; ++r)
                out[(long)(mrow + r) * K_OUT + k] = acc[ms][c][r] + bv;
        }
    }
}

extern "C" void kernel_launch(void* const* d_in, const int* in_sizes, int n_in,
                              void* d_out, int out_size, void* d_ws, size_t ws_size,
                              hipStream_t stream) {
    const float* x      = (const float*)d_in[0];
    const int*   Wq     = (const int*)  d_in[1];
    const float* scales = (const float*)d_in[2];
    const float* zeros  = (const float*)d_in[3];
    const float* mu1    = (const float*)d_in[4];
    const float* mu2    = (const float*)d_in[5];
    const float* bias   = (const float*)d_in[6];
    float* out = (float*)d_out;
    unsigned short* xb = (unsigned short*)d_ws;   // 256*8192 bf16 = 4 MB scratch

    xcvt_kernel<<<(M_ROWS * N_RED) / (256 * 8), 256, 0, stream>>>(x, xb);

    dim3 grid(K_OUT / 64, M_ROWS / 128);          // 128 x 2 = 256 workgroups
    qlin_kernel<<<grid, 256, 0, stream>>>(xb, Wq, scales, zeros, mu1, mu2, bias, out);
}
